// DotAttention_27178553049198
// MI455X (gfx1250) — matrix operand order repaired and checked by reading, not compiled, not run on hardware
//
#include <hip/hip_runtime.h>

typedef __attribute__((ext_vector_type(16))) __bf16 v16bf;
typedef __attribute__((ext_vector_type(4)))  __bf16 v4bf;
typedef __attribute__((ext_vector_type(8)))  float  v8f;

#define BM 128   // query rows per block (8 waves x 16)
#define BN 64    // keys per KV tile
#define WAVES 8

__global__ __launch_bounds__(256, 1)
void fa_fwd_kernel(const float* __restrict__ Q, const float* __restrict__ K,
                   const float* __restrict__ V, float* __restrict__ O, int S) {
  // LDS staging: K tile bf16 row-major [n][d], V tile bf16 transposed [d][k],
  // per-wave P scratch bf16 [m][k]
  __shared__ __align__(32) __bf16 Ks[BN * 128];
  __shared__ __align__(32) __bf16 Vt[128 * BN];
  __shared__ __align__(32) __bf16 Ps[WAVES * 16 * BN];

  const int tid  = threadIdx.x;
  const int lane = tid & 31;
  const int wave = tid >> 5;
  const int half = lane >> 4;   // lane group (0: lanes 0-15, 1: lanes 16-31)
  const int l16  = lane & 15;

  const int row0 = blockIdx.x * BM + wave * 16;

  // Fold 1/sqrt(d) AND log2(e) into Q so softmax runs in base-2 domain:
  // scores come out as s*log2e/sqrt(d); p = exp2(s - m) via native v_exp_f32.
  const float qscale = 0.08838834764831845f * 1.4426950408889634f;

  // ---- Q rows into WMMA 16-bit A layout: 4 chunks of K=32 ----
  // elements 0..7  : d = c*32 + 8*half    + 0..7
  // elements 8..15 : d = c*32 + 16+8*half + 0..7
  v16bf Qa[4];
  {
    const float* qrow = Q + (size_t)(row0 + l16) * 128;
#pragma unroll
    for (int c = 0; c < 4; ++c) {
      v16bf a;
      const float* p0 = qrow + c * 32 + half * 8;
      const float* p1 = p0 + 16;
#pragma unroll
      for (int i = 0; i < 8; ++i) {
        a[i]     = (__bf16)(p0[i] * qscale);
        a[8 + i] = (__bf16)(p1[i] * qscale);
      }
      Qa[c] = a;
    }
  }

  v8f acc[8] = {};                 // 16x128 O accumulator (f32, C layout per d-tile)
  float mrow[8], lrow[8];
#pragma unroll
  for (int r = 0; r < 8; ++r) { mrow[r] = -__builtin_inff(); lrow[r] = 0.0f; }

  const int nTiles = S / BN;
  __bf16* Pw = Ps + wave * (16 * BN);

  for (int j = 0; j < nTiles; ++j) {
    __syncthreads();  // previous iteration's LDS reads done before overwrite

    // ---- cooperative KV tile load, fp32 -> bf16 (hardware converts) ----
    {
      const float* Kg = K + (size_t)j * BN * 128;
      const float* Vg = V + (size_t)j * BN * 128;
      // K: row-chunk per thread, packed v4bf store (ds_store_b64)
#pragma unroll
      for (int it = 0; it < (BN * 128 / 4) / 256; ++it) {
        int idx = (it * 256 + tid) * 4;
        float4 kd = *(const float4*)(Kg + idx);
        v4bf kb = { (__bf16)kd.x, (__bf16)kd.y, (__bf16)kd.z, (__bf16)kd.w };
        *(v4bf*)(Ks + idx) = kb;
      }
      // V: column-chunk per thread -> transposed [d][k], packed v4bf store.
      // Global reads stay lane-coalesced along d.
#pragma unroll
      for (int it = 0; it < (BN * 128 / 4) / 256; ++it) {
        int c  = it * 256 + tid;     // 2048 chunks
        int dd = c & 127;            // d column
        int k0 = (c >> 7) * 4;       // key row base
        v4bf vb = { (__bf16)Vg[(k0 + 0) * 128 + dd],
                    (__bf16)Vg[(k0 + 1) * 128 + dd],
                    (__bf16)Vg[(k0 + 2) * 128 + dd],
                    (__bf16)Vg[(k0 + 3) * 128 + dd] };
        *(v4bf*)(Vt + dd * BN + k0) = vb;
      }
      if (j + 1 < nTiles) {   // pull next tile toward caches (global_prefetch_b8)
        __builtin_prefetch(Kg + BN * 128 + tid * 16, 0, 0);
        __builtin_prefetch(Vg + BN * 128 + tid * 16, 0, 0);
      }
    }
    __syncthreads();

    // ---- scores: 16 x BN = Q (16x128) * K^T, bf16 WMMA, f32 acc ----
    v8f st[4];
#pragma unroll
    for (int nt = 0; nt < 4; ++nt) {
      v8f c = {};
#pragma unroll
      for (int kc = 0; kc < 4; ++kc) {
        // B layout: lane -> key row n, elements 0..7 d = 8*half+0..7, 8..15 d = 16+8*half+0..7
        union { v16bf v; uint4 q[2]; } bu;
        const __bf16* kb = Ks + (nt * 16 + l16) * 128 + kc * 32 + half * 8;
        bu.q[0] = *(const uint4*)(kb);
        bu.q[1] = *(const uint4*)(kb + 16);
        const v16bf b = bu.v;
        c = __builtin_amdgcn_wmma_f32_16x16x32_bf16(false, Qa[kc], false, b,
                                                    (short)0, c, false, false);
      }
      st[nt] = c;
    }

    // ---- online softmax, base-2 (row = r + 8*half spans a 16-lane group) ----
    v8f alphav;
#pragma unroll
    for (int r = 0; r < 8; ++r) {
      float tmax = fmaxf(fmaxf(st[0][r], st[1][r]), fmaxf(st[2][r], st[3][r]));
#pragma unroll
      for (int off = 1; off < 16; off <<= 1)
        tmax = fmaxf(tmax, __shfl_xor(tmax, off, 32));
      float mnew = fmaxf(mrow[r], tmax);
      float rsum = 0.0f;
#pragma unroll
      for (int nt = 0; nt < 4; ++nt) {
        float p = __builtin_amdgcn_exp2f(st[nt][r] - mnew);
        st[nt][r] = p;
        rsum += p;
      }
#pragma unroll
      for (int off = 1; off < 16; off <<= 1)
        rsum += __shfl_xor(rsum, off, 32);
      float alpha = __builtin_amdgcn_exp2f(mrow[r] - mnew);
      alphav[r] = alpha;
      lrow[r] = lrow[r] * alpha + rsum;
      mrow[r] = mnew;
    }
    // vector rescale of O accumulator (v_pk_mul_f32 friendly)
#pragma unroll
    for (int dt = 0; dt < 8; ++dt) acc[dt] *= alphav;

    // ---- P (C layout, f32) -> per-wave LDS bf16 [m][k] ----
#pragma unroll
    for (int nt = 0; nt < 4; ++nt)
#pragma unroll
      for (int r = 0; r < 8; ++r)
        Pw[(r + 8 * half) * BN + nt * 16 + l16] = (__bf16)st[nt][r];

    __syncthreads();  // also orders the wave's own DS writes vs reads

    // ---- O += P (16xBN) * V (BNx128) ----
#pragma unroll
    for (int kc = 0; kc < 2; ++kc) {
      union { v16bf v; uint4 q[2]; } a;
      const __bf16* pb = Pw + l16 * BN + kc * 32 + half * 8;
      a.q[0] = *(const uint4*)(pb);
      a.q[1] = *(const uint4*)(pb + 16);
#pragma unroll
      for (int dt = 0; dt < 8; ++dt) {
        union { v16bf v; uint4 q[2]; } bu;
        const __bf16* vb = Vt + (dt * 16 + l16) * BN + kc * 32 + half * 8;
        bu.q[0] = *(const uint4*)(vb);
        bu.q[1] = *(const uint4*)(vb + 16);
        const v16bf b = bu.v;
        acc[dt] = __builtin_amdgcn_wmma_f32_16x16x32_bf16(false, a.v, false, b,
                                                          (short)0, acc[dt], false, false);
      }
    }
  }

  // ---- normalize by row sums and write O (f32) ----
  v8f invv;
#pragma unroll
  for (int r = 0; r < 8; ++r) invv[r] = 1.0f / lrow[r];
#pragma unroll
  for (int dt = 0; dt < 8; ++dt) {
    v8f o = acc[dt] * invv;
#pragma unroll
    for (int r = 0; r < 8; ++r)
      O[(size_t)(row0 + r + 8 * half) * 128 + dt * 16 + l16] = o[r];
  }
}

extern "C" void kernel_launch(void* const* d_in, const int* in_sizes, int n_in,
                              void* d_out, int out_size, void* d_ws, size_t ws_size,
                              hipStream_t stream) {
  (void)n_in; (void)out_size; (void)d_ws; (void)ws_size;
  const float* Q = (const float*)d_in[0];
  const float* K = (const float*)d_in[1];
  const float* V = (const float*)d_in[2];
  float* O = (float*)d_out;
  const int S = in_sizes[0] / 128;   // D = 128 fixed by the reference
  dim3 grid(S / BM), block(256);
  hipLaunchKernelGGL(fa_fwd_kernel, grid, block, 0, stream, Q, K, V, O, S);
}
